// Transformer_22016002359541
// MI455X (gfx1250) — matrix operand order repaired and checked by reading, not compiled, not run on hardware
//
#include <hip/hip_runtime.h>
#include <hip/hip_bf16.h>

typedef __attribute__((ext_vector_type(16))) _Float16 v16h;
typedef __attribute__((ext_vector_type(8)))  _Float16 v8h;
typedef __attribute__((ext_vector_type(8)))  float    v8f;

#define B_  32
#define S_  1024
#define IN_ 16
#define E_  128
#define NH_ 4
#define DH_ 32
#define QKVLD_ 384   // 3*E

// ---------------------------------------------------------------------------
// WMMA helpers (CDNA5 wave32 layouts, ISA 7.12.2)
// ---------------------------------------------------------------------------
__device__ inline v8f wmma_f32(v16h a, v16h b, v8f c) {
  // (neg_a, A, neg_b, B, c_mod, C, reuse_a, reuse_b)
  return __builtin_amdgcn_wmma_f32_16x16x32_f16(false, a, false, b, (short)0, c,
                                                false, false);
}

// A-matrix 16x32 f16 tile from row-major [*, ld], rows row0..+15, cols k0..+31.
// lane<16: row=lane, K = {k0..k0+15}
// lane>=16: row=lane-16, K = {k0+16..k0+31}
__device__ inline v16h load_A16x32(const _Float16* __restrict__ base, int ld,
                                   int row0, int k0) {
  int l = threadIdx.x & 31;
  const _Float16* p = base + (size_t)(row0 + (l & 15)) * ld + k0 + ((l < 16) ? 0 : 16);
  v8h lo = *(const v8h*)p;          // 8 contiguous halfs, 16B aligned
  v8h hi = *(const v8h*)(p + 8);
  v16h a;
#pragma unroll
  for (int i = 0; i < 8; i++) { a[i] = lo[i]; a[i + 8] = hi[i]; }
  return a;
}

// B-matrix 32x16 (KxN) built as W^T from row-major W[N,K] (ld = K stride).
// lane -> column n0+(l&15); K values = k0 + (l<16?0:16) + i, i=0..15 contiguous.
__device__ inline v16h load_BT32x16(const _Float16* __restrict__ base, int ld,
                                    int n0, int k0) {
  int l = threadIdx.x & 31;
  const _Float16* p = base + (size_t)(n0 + (l & 15)) * ld + k0 + ((l < 16) ? 0 : 16);
  v8h lo = *(const v8h*)p;
  v8h hi = *(const v8h*)(p + 8);
  v16h b;
#pragma unroll
  for (int i = 0; i < 8; i++) { b[i] = lo[i]; b[i + 8] = hi[i]; }
  return b;
}

// ---------------------------------------------------------------------------
// fp32 -> f16 weight conversion
// ---------------------------------------------------------------------------
__global__ void cvt_f16_kernel(const float* __restrict__ src,
                               _Float16* __restrict__ dst, int n) {
  int i = blockIdx.x * blockDim.x + threadIdx.x;
  if (i < n) dst[i] = (_Float16)src[i];
}

// ---------------------------------------------------------------------------
// Embedding: h = relu(x @ Wᵀ + b), x[M,16], W[128,16]  (K=16, VALU fp32)
// ---------------------------------------------------------------------------
__global__ void embed_kernel(const float* __restrict__ x,
                             const float* __restrict__ w,
                             const float* __restrict__ b,
                             float* __restrict__ h32,
                             _Float16* __restrict__ h16, int M) {
  int i = blockIdx.x * blockDim.x + threadIdx.x;
  if (i >= M * E_) return;
  int m = i >> 7, e = i & 127;
  const float* xr = x + (size_t)m * IN_;
  const float* wr = w + (size_t)e * IN_;
  float s = b[e];
#pragma unroll
  for (int k = 0; k < IN_; k++) s += xr[k] * wr[k];
  s = fmaxf(s, 0.f);
  h32[i] = s;
  h16[i] = (_Float16)s;
}

// ---------------------------------------------------------------------------
// GEMM: Y[M,N] = act(X[M,K]f16 @ W[N,K]ᵀ f16 + bias)
// One wave computes a 16x64 strip (4 accumulators): A tile loaded once per
// K-step and reused for 4 WMMAs -> 4x less A traffic, denser wmma mix.
// Requires N % 64 == 0.
// ---------------------------------------------------------------------------
__global__ void gemm_wmma_kernel(const _Float16* __restrict__ X,
                                 const _Float16* __restrict__ W,
                                 const float* __restrict__ bias,
                                 float* __restrict__ Yf32,
                                 _Float16* __restrict__ Yf16,
                                 int M, int N, int K, int relu) {
  int wave = (int)((blockIdx.x * blockDim.x + threadIdx.x) >> 5);
  int sN = N >> 6;                        // 64-wide column strips
  int total = (M >> 4) * sN;
  if (wave >= total) return;              // uniform per-wave exit: EXEC full
  int tm = wave / sN, tn = wave - tm * sN;
  int row0 = tm << 4, n0 = tn << 6;

  v8f acc[4] = {{}, {}, {}, {}};
  for (int k0 = 0; k0 < K; k0 += 32) {
    v16h a = load_A16x32(X, K, row0, k0);
#pragma unroll
    for (int t = 0; t < 4; t++) {
      v16h b = load_BT32x16(W, K, n0 + 16 * t, k0);
      acc[t] = wmma_f32(a, b, acc[t]);
    }
  }

  int l = threadIdx.x & 31;
  int rbase = row0 + ((l < 16) ? 0 : 8);  // D layout: VGPR j -> row rbase+j
#pragma unroll
  for (int t = 0; t < 4; t++) {
    int col = n0 + 16 * t + (l & 15);
    float bn = bias ? bias[col] : 0.f;
#pragma unroll
    for (int j = 0; j < 8; j++) {
      float v = acc[t][j] + bn;
      if (relu) v = fmaxf(v, 0.f);
      size_t idx = (size_t)(rbase + j) * N + col;
      if (Yf32) Yf32[idx] = v;
      if (Yf16) Yf16[idx] = (_Float16)v;
    }
  }
}

// ---------------------------------------------------------------------------
// Fused flash attention: one wave per (b, head, 16-row q tile), KV tiles of 32
// qkv: [B,S,3E] f16 ; ctx out: [B,S,E] f16
// V tile staged through LDS with coalesced b128 loads (column-major B-operand
// gather happens in LDS, not scattered u16 global loads).
// ---------------------------------------------------------------------------
__global__ void attn_kernel(const _Float16* __restrict__ qkv,
                            _Float16* __restrict__ ctx) {
  __shared__ __align__(16) _Float16 pbuf[4][16 * 32];  // per-wave P staging
  __shared__ __align__(16) _Float16 vbuf[4][32 * 32];  // per-wave V tile
  int wid = threadIdx.x >> 5;
  int l = threadIdx.x & 31;
  int gw = blockIdx.x * 4 + wid;        // 0 .. B*NH*(S/16)-1 = 8191
  int qt = gw & 63;                     // S/16 = 64
  int bh = gw >> 6;
  int h = bh & (NH_ - 1);
  int b = bh >> 2;

  const _Float16* qb = qkv + (size_t)b * S_ * QKVLD_ + h * DH_;
  const _Float16* kb = qb + E_;
  const _Float16* vb = qb + 2 * E_;
  int row0 = qt << 4;

  v16h Aq = load_A16x32(qb, QKVLD_, row0, 0);   // Q tile 16 x DH(=32)

  float mrow[8], lrow[8];
#pragma unroll
  for (int j = 0; j < 8; j++) { mrow[j] = -1e30f; lrow[j] = 0.f; }
  v8f acc0 = {}, acc1 = {};
  const float scale = 0.17677669529663687f;     // 1/sqrt(DH)
  _Float16* pl = pbuf[wid];
  _Float16* vl = vbuf[wid];
  int rb = (l < 16) ? 0 : 8;
  int c = l & 15;

  for (int kt = 0; kt < S_; kt += 32) {
    // --- coalesced V tile load: lane l copies kv-row kt+l (32 halfs) ---
    {
      const _Float16* vr = vb + (size_t)(kt + l) * QKVLD_;
      v8h r0 = *(const v8h*)vr;
      v8h r1 = *(const v8h*)(vr + 8);
      v8h r2 = *(const v8h*)(vr + 16);
      v8h r3 = *(const v8h*)(vr + 24);
      *(v8h*)(vl + l * 32)      = r0;
      *(v8h*)(vl + l * 32 + 8)  = r1;
      *(v8h*)(vl + l * 32 + 16) = r2;
      *(v8h*)(vl + l * 32 + 24) = r3;
    }

    v16h B0 = load_BT32x16(kb, QKVLD_, kt, 0);       // Kᵀ cols kt..kt+15
    v16h B1 = load_BT32x16(kb, QKVLD_, kt + 16, 0);  // cols kt+16..kt+31
    v8f z = {};
    v8f s0 = wmma_f32(Aq, B0, z);
    v8f s1 = wmma_f32(Aq, B1, z);

    float p0[8], p1[8];
#pragma unroll
    for (int j = 0; j < 8; j++) {
      float x0 = s0[j] * scale, x1 = s1[j] * scale;
      float mx = fmaxf(x0, x1);
#pragma unroll
      for (int msk = 8; msk >= 1; msk >>= 1)       // reduce within 16-lane half
        mx = fmaxf(mx, __shfl_xor(mx, msk, 32));
      float mnew = fmaxf(mrow[j], mx);
      float alpha = __expf(mrow[j] - mnew);
      mrow[j] = mnew;
      float e0 = __expf(x0 - mnew), e1 = __expf(x1 - mnew);
      float rs = e0 + e1;
#pragma unroll
      for (int msk = 8; msk >= 1; msk >>= 1)
        rs += __shfl_xor(rs, msk, 32);
      lrow[j] = lrow[j] * alpha + rs;
      acc0[j] *= alpha;
      acc1[j] *= alpha;
      p0[j] = e0; p1[j] = e1;
    }

    // D layout -> row-major 16x32 P in LDS
#pragma unroll
    for (int j = 0; j < 8; j++) {
      pl[(j + rb) * 32 + c] = (_Float16)p0[j];
      pl[(j + rb) * 32 + 16 + c] = (_Float16)p1[j];
    }
    __syncthreads();

    v16h Ap = load_A16x32(pl, 32, 0, 0);  // P as A operand (transposed read)

    // V B-tiles from LDS: B(k,n) = V[kt+k][n]; lane -> column n, K strided
    v16h Bv0, Bv1;
    {
      int kbase = (l < 16) ? 0 : 16;
      const _Float16* vp = vl + kbase * 32 + c;
#pragma unroll
      for (int i = 0; i < 16; i++) {
        Bv0[i] = vp[i * 32];        // d = c      (cols 0..15)
        Bv1[i] = vp[i * 32 + 16];   // d = 16 + c (cols 16..31)
      }
    }
    __syncthreads();

    acc0 = wmma_f32(Ap, Bv0, acc0);
    acc1 = wmma_f32(Ap, Bv1, acc1);
  }

  _Float16* ob = ctx + ((size_t)b * S_ + row0) * E_ + h * DH_;
#pragma unroll
  for (int j = 0; j < 8; j++) {
    float inv = 1.f / lrow[j];
    ob[(size_t)(j + rb) * E_ + c] = (_Float16)(acc0[j] * inv);
    ob[(size_t)(j + rb) * E_ + 16 + c] = (_Float16)(acc1[j] * inv);
  }
}

// ---------------------------------------------------------------------------
// Residual add + LayerNorm over E=128; one wave per row (lane covers 4 elems)
// ---------------------------------------------------------------------------
__global__ void add_ln_kernel(const float* __restrict__ a,   // projection out
                              const float* __restrict__ r,   // residual stream
                              const float* __restrict__ g,
                              const float* __restrict__ bt,
                              float* __restrict__ h32,
                              _Float16* __restrict__ h16, int M) {
  int wave = (int)((blockIdx.x * blockDim.x + threadIdx.x) >> 5);
  if (wave >= M) return;
  int l = threadIdx.x & 31;
  const float* ar = a + (size_t)wave * E_;
  const float* rr = r + (size_t)wave * E_;
  float v[4];
  float s = 0.f;
#pragma unroll
  for (int i = 0; i < 4; i++) {
    v[i] = ar[l + 32 * i] + rr[l + 32 * i];
    s += v[i];
  }
#pragma unroll
  for (int m = 16; m >= 1; m >>= 1) s += __shfl_xor(s, m, 32);
  float mu = s * (1.f / 128.f);
  float var = 0.f;
#pragma unroll
  for (int i = 0; i < 4; i++) { float d = v[i] - mu; var += d * d; }
#pragma unroll
  for (int m = 16; m >= 1; m >>= 1) var += __shfl_xor(var, m, 32);
  float inv = rsqrtf(var * (1.f / 128.f) + 1e-5f);
#pragma unroll
  for (int i = 0; i < 4; i++) {
    int e = l + 32 * i;
    float o = (v[i] - mu) * inv * g[e] + bt[e];
    h32[(size_t)wave * E_ + e] = o;
    h16[(size_t)wave * E_ + e] = (_Float16)o;
  }
}

// ---------------------------------------------------------------------------
// Masked sum pooling: pooled[b,e] = sum_s h[b,s,e]*mask[b,s]
// ---------------------------------------------------------------------------
__global__ void pool_kernel(const float* __restrict__ h,
                            const float* __restrict__ mask,
                            float* __restrict__ pooled) {
  int i = blockIdx.x * blockDim.x + threadIdx.x;
  if (i >= B_ * E_) return;
  int b = i >> 7, e = i & 127;
  float s = 0.f;
  for (int t = 0; t < S_; t++)
    s += h[((size_t)b * S_ + t) * E_ + e] * mask[b * S_ + t];
  pooled[i] = s;
}

// ---------------------------------------------------------------------------
// Small classifier FC: act=0 none, 1 relu, 2 sigmoid (fp32 naive, M=32)
// ---------------------------------------------------------------------------
__global__ void fc_kernel(const float* __restrict__ in,
                          const float* __restrict__ w,
                          const float* __restrict__ b,
                          float* __restrict__ out, int M, int N, int K, int act) {
  int i = blockIdx.x * blockDim.x + threadIdx.x;
  if (i >= M * N) return;
  int m = i / N, n = i - m * N;
  float s = b[n];
  for (int k = 0; k < K; k++) s += in[(size_t)m * K + k] * w[(size_t)n * K + k];
  if (act == 1) s = fmaxf(s, 0.f);
  if (act == 2) s = 1.f / (1.f + __expf(-s));
  out[i] = s;
}

// ---------------------------------------------------------------------------
extern "C" void kernel_launch(void* const* d_in, const int* in_sizes, int n_in,
                              void* d_out, int out_size, void* d_ws, size_t ws_size,
                              hipStream_t stream) {
  const float* x        = (const float*)d_in[0];
  const float* mask     = (const float*)d_in[1];
  const float* embed_w  = (const float*)d_in[2];
  const float* embed_b  = (const float*)d_in[3];
  const float* in_proj_w= (const float*)d_in[4];
  const float* in_proj_b= (const float*)d_in[5];
  const float* out_w    = (const float*)d_in[6];
  const float* out_b    = (const float*)d_in[7];
  const float* ln1_g    = (const float*)d_in[8];
  const float* ln1_b    = (const float*)d_in[9];
  const float* ln2_g    = (const float*)d_in[10];
  const float* ln2_b    = (const float*)d_in[11];
  const float* ffn_w1   = (const float*)d_in[12];
  const float* ffn_b1   = (const float*)d_in[13];
  const float* ffn_w2   = (const float*)d_in[14];
  const float* ffn_b2   = (const float*)d_in[15];
  const float* cls_w1   = (const float*)d_in[16];
  const float* cls_b1   = (const float*)d_in[17];
  const float* cls_w2   = (const float*)d_in[18];
  const float* cls_b2   = (const float*)d_in[19];
  const float* cls_w3   = (const float*)d_in[20];
  const float* cls_b3   = (const float*)d_in[21];
  const float* cls_w4   = (const float*)d_in[22];
  const float* cls_b4   = (const float*)d_in[23];

  const int M = B_ * S_;        // 32768
  char* ws = (char*)d_ws;
  size_t off = 0;
  auto alloc = [&](size_t bytes) -> char* {
    char* p = ws + off;
    off = (off + bytes + 255) & ~(size_t)255;
    return p;
  };

  _Float16* wq16 = (_Float16*)alloc((size_t)3 * 384 * 128 * 2);
  _Float16* wo16 = (_Float16*)alloc((size_t)3 * 128 * 128 * 2);
  _Float16* w116 = (_Float16*)alloc((size_t)3 * 128 * 128 * 2);
  _Float16* w216 = (_Float16*)alloc((size_t)3 * 128 * 128 * 2);
  float*    h32  = (float*)   alloc((size_t)M * E_ * 4);
  _Float16* h16  = (_Float16*)alloc((size_t)M * E_ * 2);
  _Float16* qkv  = (_Float16*)alloc((size_t)M * QKVLD_ * 2);
  _Float16* ctx  = (_Float16*)alloc((size_t)M * E_ * 2);
  _Float16* ffh  = (_Float16*)alloc((size_t)M * E_ * 2);
  float*    tmp  = (float*)   alloc((size_t)M * E_ * 4);
  float*    pooled = (float*) alloc((size_t)B_ * E_ * 4);
  float*    z1   = (float*)   alloc((size_t)B_ * 256 * 4);
  float*    z2   = (float*)   alloc((size_t)B_ * 256 * 4);
  float*    z3   = (float*)   alloc((size_t)B_ * 256 * 4);
  (void)ws_size; (void)n_in; (void)in_sizes; (void)out_size;

  auto cdiv = [](long a, long b) { return (a + b - 1) / b; };
  auto cvt = [&](const float* s, _Float16* d, int n) {
    cvt_f16_kernel<<<cdiv(n, 256), 256, 0, stream>>>(s, d, n);
  };
  auto gemm = [&](const _Float16* X, const _Float16* W, const float* bias,
                  float* y32, _Float16* y16, int m, int n, int k, int relu) {
    long waves = (long)(m / 16) * (n / 64);   // 16x64 strip per wave
    gemm_wmma_kernel<<<cdiv(waves * 32, 256), 256, 0, stream>>>(
        X, W, bias, y32, y16, m, n, k, relu);
  };

  // Weight conversion (all 3 layers at once per tensor)
  cvt(in_proj_w, wq16, 3 * 384 * 128);
  cvt(out_w,     wo16, 3 * 128 * 128);
  cvt(ffn_w1,    w116, 3 * 128 * 128);
  cvt(ffn_w2,    w216, 3 * 128 * 128);

  // Embedding
  embed_kernel<<<cdiv((long)M * E_, 256), 256, 0, stream>>>(
      x, embed_w, embed_b, h32, h16, M);

  // 3 encoder layers
  for (int l = 0; l < 3; l++) {
    gemm(h16, wq16 + (size_t)l * 384 * 128, in_proj_b + l * 384,
         nullptr, qkv, M, 384, 128, 0);

    // B*NH*(S/16) = 8192 waves, 4 waves/block
    attn_kernel<<<2048, 128, 0, stream>>>(qkv, ctx);

    gemm(ctx, wo16 + (size_t)l * 128 * 128, out_b + l * 128,
         tmp, nullptr, M, 128, 128, 0);
    add_ln_kernel<<<cdiv((long)M * 32, 256), 256, 0, stream>>>(
        tmp, h32, ln1_g + l * 128, ln1_b + l * 128, h32, h16, M);

    gemm(h16, w116 + (size_t)l * 128 * 128, ffn_b1 + l * 128,
         nullptr, ffh, M, 128, 128, 1);
    gemm(ffh, w216 + (size_t)l * 128 * 128, ffn_b2 + l * 128,
         tmp, nullptr, M, 128, 128, 0);
    add_ln_kernel<<<cdiv((long)M * 32, 256), 256, 0, stream>>>(
        tmp, h32, ln2_g + l * 128, ln2_b + l * 128, h32, h16, M);
  }

  // Pool + classifier
  pool_kernel<<<cdiv((long)B_ * E_, 256), 256, 0, stream>>>(h32, mask, pooled);
  fc_kernel<<<cdiv((long)B_ * 256, 256), 256, 0, stream>>>(pooled, cls_w1, cls_b1, z1, B_, 256, 128, 1);
  fc_kernel<<<cdiv((long)B_ * 256, 256), 256, 0, stream>>>(z1, cls_w2, cls_b2, z2, B_, 256, 256, 1);
  fc_kernel<<<cdiv((long)B_ * 256, 256), 256, 0, stream>>>(z2, cls_w3, cls_b3, z3, B_, 256, 256, 1);
  fc_kernel<<<cdiv((long)B_, 256), 256, 0, stream>>>(z3, cls_w4, cls_b4, (float*)d_out, B_, 1, 256, 2);
}
